// HypercomplexMoE_73375221284955
// MI455X (gfx1250) — compile-verified, matrix operand order checked by reading, not run on hardware
//
#include <hip/hip_runtime.h>
#include <hip/hip_bf16.h>
#include <math.h>

// ---------------------------------------------------------------------------
// HypercomplexMoE for MI455X (gfx1250)
//   - bf16 WMMA GEMMs (v_wmma_f32_16x16x32_bf16), fp32 accumulate
//   - Tensor Data Mover (tensor_load_to_lds) fills double-buffered LDS tiles,
//     with TDM LDS-padding (144B row pitch) to avoid bank conflicts
//   - TENSORcnt / workgroup-barrier pipeline
// ---------------------------------------------------------------------------

typedef __attribute__((ext_vector_type(16))) __bf16 v16bf;
typedef __attribute__((ext_vector_type(8)))  float  v8f;
typedef unsigned int u32x4 __attribute__((ext_vector_type(4)));
typedef int          i32x4 __attribute__((ext_vector_type(4)));
typedef int          i32x8 __attribute__((ext_vector_type(8)));

static constexpr int kTok = 8192;   // B*N = 8*1024
static constexpr int kC   = 768;
static constexpr int kE   = 8;
static constexpr int kDFF = 3072;

static constexpr int KTILE  = 64;   // K elements staged per LDS tile
static constexpr int PITCH  = 72;   // LDS row pitch in halfs (64 data + 4-DWORD TDM pad)

union Frag {
    v16bf v;
    uint4 q[2];
};

__device__ __forceinline__ unsigned short f2bf_bits(float f) {
    union { float f; unsigned u; } v; v.f = f;
    unsigned r = (v.u + 0x7FFFu + ((v.u >> 16) & 1u)) >> 16;
    return (unsigned short)r;
}

// --------------------------- TDM helper ------------------------------------
// 2D tile load: rows x KTILE bf16 elements, row-major source with stride
// lda (elements). LDS destination padded: every 32 DWORDs (128B = one row)
// insert 4 DWORDs (16B) -> 144B row pitch.
__device__ __forceinline__ void tdm_load_tile(const unsigned short* gsrc,
                                              unsigned lds_off, int rows, int lda) {
    unsigned long long ga = (unsigned long long)(uintptr_t)gsrc;
    u32x4 g0 = { 1u,                                        // count=1 (valid D#)
                 lds_off,                                   // lds_addr (bytes)
                 (unsigned)(ga & 0xFFFFFFFFull),            // global_addr[31:0]
                 (unsigned)((ga >> 32) & 0x01FFFFFFull)     // global_addr[56:32]
                     | (2u << 30) };                        // type=2 (image)
    i32x8 g1 = { (int)((1u << 16)          // data_size = 1 -> 2 bytes
                     | (1u << 20)          // pad_enable
                     | (4u << 22)          // pad_interval: 32 DWORDs (128B)
                     | (3u << 25)),        // pad_amount:   4 DWORDs (16B)
                 (int)((unsigned)KTILE << 16),      // tensor_dim0[15:0]  (bits 63:48)
                 (int)((unsigned)rows  << 16),      // dim0 hi=0 | tensor_dim1 lo
                 (int)((unsigned)KTILE << 16),      // dim1 hi=0 | tile_dim0
                 rows,                              // tile_dim1 | tile_dim2=0
                 lda,                               // tensor_dim0_stride[31:0]
                 0, 0 };                            // stride hi / dim1_stride
    i32x4 gz = { 0, 0, 0, 0 };
#if defined(__clang_major__) && (__clang_major__ >= 23)
    i32x8 gz8 = { 0, 0, 0, 0, 0, 0, 0, 0 };
    __builtin_amdgcn_tensor_load_to_lds(g0, g1, gz, gz, gz8, 0);
#else
    __builtin_amdgcn_tensor_load_to_lds(g0, g1, gz, gz, 0);
#endif
}

// Fragment fetch from a padded LDS tile (two ds_load_b128 per fragment).
// 16-bit A-operand lane layout: lanes 0-15 row=lane, K=k0+{0..7},{16..23};
// lanes 16-31 row=lane-16, K=k0+{8..15},{24..31}.
__device__ __forceinline__ Frag lds_frag(const unsigned short* tile,
                                         int row, int kc, int khalf) {
    Frag f;
    const unsigned short* p = tile + row * PITCH + kc + khalf;
    f.q[0] = *(const uint4*)p;
    f.q[1] = *(const uint4*)(p + 16);
    return f;
}

// --------------------------- prep kernels ----------------------------------

// W_r[a*2+x, b*192+y] = sum_n rA[n,a,b] * rS[n,x,y]   -> [8,768] fp32
__global__ void k_build_wr(const float* __restrict__ rA, const float* __restrict__ rS,
                           float* __restrict__ Wr) {
    int idx = blockIdx.x * 256 + threadIdx.x;
    if (idx >= kE * kC) return;
    int row = idx / kC, col = idx % kC;
    int a = row >> 1, xo = row & 1;
    int b = col / 192, y = col % 192;
    float s = 0.f;
#pragma unroll
    for (int nn = 0; nn < 4; ++nn)
        s += rA[nn * 16 + a * 4 + b] * rS[(nn * 2 + xo) * 192 + y];
    Wr[idx] = s;
}

// W_e[e, a*768+x, b*192+y] = sum_n eA[e,n,a,b]*eS[e,n,x,y]  -> bf16 [8,3072,768]
__global__ void k_build_we(const float* __restrict__ eA, const float* __restrict__ eS,
                           unsigned short* __restrict__ We) {
    int i = blockIdx.x * 256 + threadIdx.x;
    if (i >= kE * kDFF * kC) return;
    int e   = i / (kDFF * kC);
    int rem = i % (kDFF * kC);
    int r = rem / kC, c = rem % kC;
    int a = r / 768, xo = r % 768;
    int b = c / 192, y  = c % 192;
    float s = 0.f;
#pragma unroll
    for (int nn = 0; nn < 4; ++nn)
        s += eA[e * 64 + nn * 16 + a * 4 + b] *
             eS[((size_t)(e * 4 + nn) * 768 + xo) * 192 + y];
    We[i] = f2bf_bits(s);
}

// ndWT[e,c,o] = bf16(nd_W[e,o,c])   -> bf16 [8,768,3072]
__global__ void k_transpose_ndw(const float* __restrict__ W, unsigned short* __restrict__ WT) {
    int i = blockIdx.x * 256 + threadIdx.x;
    if (i >= kE * kC * kDFF) return;
    int e   = i / (kC * kDFF);
    int rem = i % (kC * kDFF);
    int c = rem / kDFF, o = rem % kDFF;
    WT[i] = f2bf_bits(W[((size_t)e * kDFF + o) * kC + c]);
}

__global__ void k_convert_f32_bf16(const float* __restrict__ in,
                                   unsigned short* __restrict__ out, int ntot) {
    int i = blockIdx.x * 256 + threadIdx.x;
    if (i < ntot) out[i] = f2bf_bits(in[i]);
}

__global__ void k_zero_f32(float* __restrict__ p, int ntot) {
    int i = blockIdx.x * 256 + threadIdx.x;
    if (i < ntot) p[i] = 0.f;
}

// Router logits + top-2 softmax gate. One thread per token.
__global__ void k_router_gate(const float* __restrict__ x, const float* __restrict__ Wr,
                              const float* __restrict__ rb, const float* __restrict__ dr,
                              const int* __restrict__ did, float* __restrict__ gate) {
    int t = blockIdx.x * 256 + threadIdx.x;
    if (t >= kTok) return;
    int d = did[0];
    float acc[kE];
#pragma unroll
    for (int e = 0; e < kE; ++e) acc[e] = rb[e] + dr[d * kE + e];
    const float* xr = x + (size_t)t * kC;
    for (int c = 0; c < kC; ++c) {
        float xv = xr[c];
#pragma unroll
        for (int e = 0; e < kE; ++e) acc[e] += xv * Wr[e * kC + c];
    }
    int i1 = 0; float l1 = acc[0];
#pragma unroll
    for (int e = 1; e < kE; ++e) if (acc[e] > l1) { l1 = acc[e]; i1 = e; }
    int i2 = 0; float l2 = -3.0e38f;
#pragma unroll
    for (int e = 0; e < kE; ++e) if (e != i1 && acc[e] > l2) { l2 = acc[e]; i2 = e; }
    float w1 = 1.f / (1.f + expf(l2 - l1));   // softmax over {l1,l2}, l1 >= l2
    float w2 = 1.f - w1;
#pragma unroll
    for (int e = 0; e < kE; ++e) gate[t * kE + e] = 0.f;
    gate[t * kE + i1] = w1;
    gate[t * kE + i2] = w2;
}

// --------------------------- WMMA GEMM kernels -----------------------------
// 256 threads = 8 waves, block tile 64(M) x 256(N), wave tile 32x64 (2x4
// fragments). K staged in 64-deep LDS tiles filled by TDM, double buffered:
//   wave0: tensor_load_to_lds(A[next]), tensor_load_to_lds(B[next])
//   all:   16 wmmas from LDS buf[cur]
//   wave0: s_wait_tensorcnt 0 ; all: barrier ; swap

// FFN1: H[m,o] = bf16(gelu(x_bf16[m,:] . We[o,:] + ebias[o]))
__global__ __launch_bounds__(256)
void k_ffn1(const unsigned short* __restrict__ Xb,   // [8192,768]
            const unsigned short* __restrict__ We,   // [3072,768] (one expert)
            const float* __restrict__ ebias,         // [3072]
            unsigned short* __restrict__ H)          // [8192,3072]
{
    __shared__ __align__(16) unsigned short sA[2][64  * PITCH];
    __shared__ __align__(16) unsigned short sB[2][256 * PITCH];

    const int lane  = threadIdx.x & 31;
    const int wave  = threadIdx.x >> 5;
    const int Mblk  = blockIdx.y * 64;
    const int Nblk  = blockIdx.x * 256;
    const int waveM = (wave >> 2) * 32;   // 0 / 32
    const int waveN = (wave & 3) * 64;    // 0..192
    const int lrow  = lane & 15;
    const int khalf = (lane >> 4) << 3;   // 0 or 8
    const v8f zero = {0.f, 0.f, 0.f, 0.f, 0.f, 0.f, 0.f, 0.f};

    v8f acc[2][4];
#pragma unroll
    for (int f = 0; f < 2; ++f)
#pragma unroll
        for (int g = 0; g < 4; ++g) acc[f][g] = zero;

    const int STAGES = kC / KTILE;   // 12
    if (wave == 0) {
        tdm_load_tile(Xb + (size_t)Mblk * kC, (unsigned)(uintptr_t)&sA[0][0], 64,  kC);
        tdm_load_tile(We + (size_t)Nblk * kC, (unsigned)(uintptr_t)&sB[0][0], 256, kC);
        __builtin_amdgcn_s_wait_tensorcnt(0);
    }
    __syncthreads();

    for (int s = 0; s < STAGES; ++s) {
        const int cur = s & 1, nxt = (s + 1) & 1;
        if (wave == 0 && s + 1 < STAGES) {
            tdm_load_tile(Xb + (size_t)Mblk * kC + (s + 1) * KTILE,
                          (unsigned)(uintptr_t)&sA[nxt][0], 64, kC);
            tdm_load_tile(We + (size_t)Nblk * kC + (s + 1) * KTILE,
                          (unsigned)(uintptr_t)&sB[nxt][0], 256, kC);
        }
#pragma unroll
        for (int kc = 0; kc < KTILE; kc += 32) {
            Frag af[2], bfr[4];
#pragma unroll
            for (int f = 0; f < 2; ++f)
                af[f] = lds_frag(&sA[cur][0], waveM + f * 16 + lrow, kc, khalf);
#pragma unroll
            for (int g = 0; g < 4; ++g)
                bfr[g] = lds_frag(&sB[cur][0], waveN + g * 16 + lrow, kc, khalf);
#pragma unroll
            for (int f = 0; f < 2; ++f)
#pragma unroll
                for (int g = 0; g < 4; ++g)
                    acc[f][g] = __builtin_amdgcn_wmma_f32_16x16x32_bf16(
                        false, af[f].v, false, bfr[g].v, (short)0, acc[f][g], false, false);
        }
        if (wave == 0 && s + 1 < STAGES) __builtin_amdgcn_s_wait_tensorcnt(0);
        __syncthreads();
    }

    // epilogue: bias + exact GELU + bf16 store
#pragma unroll
    for (int f = 0; f < 2; ++f) {
#pragma unroll
        for (int r = 0; r < 8; ++r) {
            int row = Mblk + waveM + f * 16 + r + ((lane >> 4) << 3);
#pragma unroll
            for (int g = 0; g < 4; ++g) {
                int col = Nblk + waveN + g * 16 + lrow;
                float v = acc[f][g][r] + ebias[col];
                v = 0.5f * v * (1.0f + erff(v * 0.70710678118654752f));
                H[(size_t)row * kDFF + col] = f2bf_bits(v);
            }
        }
    }
}

// FFN2: out[m,c] += gate[m,e] * (H[m,:] . ndWT[c,:] + nb[c])
__global__ __launch_bounds__(256)
void k_ffn2(const unsigned short* __restrict__ H,    // [8192,3072]
            const unsigned short* __restrict__ WT,   // [768,3072] (nd_W[e]^T)
            const float* __restrict__ nb,            // [768]
            const float* __restrict__ gate,          // [8192,8]
            int e,
            float* __restrict__ out)                 // [8192,768] accumulated
{
    __shared__ __align__(16) unsigned short sA[2][64  * PITCH];
    __shared__ __align__(16) unsigned short sB[2][256 * PITCH];

    const int lane  = threadIdx.x & 31;
    const int wave  = threadIdx.x >> 5;
    const int Mblk  = blockIdx.y * 64;
    const int Nblk  = blockIdx.x * 256;
    const int waveM = (wave >> 2) * 32;
    const int waveN = (wave & 3) * 64;
    const int lrow  = lane & 15;
    const int khalf = (lane >> 4) << 3;
    const v8f zero = {0.f, 0.f, 0.f, 0.f, 0.f, 0.f, 0.f, 0.f};

    v8f acc[2][4];
#pragma unroll
    for (int f = 0; f < 2; ++f)
#pragma unroll
        for (int g = 0; g < 4; ++g) acc[f][g] = zero;

    const int STAGES = kDFF / KTILE;   // 48
    if (wave == 0) {
        tdm_load_tile(H  + (size_t)Mblk * kDFF, (unsigned)(uintptr_t)&sA[0][0], 64,  kDFF);
        tdm_load_tile(WT + (size_t)Nblk * kDFF, (unsigned)(uintptr_t)&sB[0][0], 256, kDFF);
        __builtin_amdgcn_s_wait_tensorcnt(0);
    }
    __syncthreads();

    for (int s = 0; s < STAGES; ++s) {
        const int cur = s & 1, nxt = (s + 1) & 1;
        if (wave == 0 && s + 1 < STAGES) {
            tdm_load_tile(H  + (size_t)Mblk * kDFF + (s + 1) * KTILE,
                          (unsigned)(uintptr_t)&sA[nxt][0], 64, kDFF);
            tdm_load_tile(WT + (size_t)Nblk * kDFF + (s + 1) * KTILE,
                          (unsigned)(uintptr_t)&sB[nxt][0], 256, kDFF);
        }
#pragma unroll
        for (int kc = 0; kc < KTILE; kc += 32) {
            Frag af[2], bfr[4];
#pragma unroll
            for (int f = 0; f < 2; ++f)
                af[f] = lds_frag(&sA[cur][0], waveM + f * 16 + lrow, kc, khalf);
#pragma unroll
            for (int g = 0; g < 4; ++g)
                bfr[g] = lds_frag(&sB[cur][0], waveN + g * 16 + lrow, kc, khalf);
#pragma unroll
            for (int f = 0; f < 2; ++f)
#pragma unroll
                for (int g = 0; g < 4; ++g)
                    acc[f][g] = __builtin_amdgcn_wmma_f32_16x16x32_bf16(
                        false, af[f].v, false, bfr[g].v, (short)0, acc[f][g], false, false);
        }
        if (wave == 0 && s + 1 < STAGES) __builtin_amdgcn_s_wait_tensorcnt(0);
        __syncthreads();
    }

    // epilogue: bias + gate-weighted accumulation into fp32 output
#pragma unroll
    for (int f = 0; f < 2; ++f) {
#pragma unroll
        for (int r = 0; r < 8; ++r) {
            int row = Mblk + waveM + f * 16 + r + ((lane >> 4) << 3);
            float gr = gate[row * kE + e];
#pragma unroll
            for (int g = 0; g < 4; ++g) {
                int col = Nblk + waveN + g * 16 + lrow;
                out[(size_t)row * kC + col] += gr * (acc[f][g][r] + nb[col]);
            }
        }
    }
}

// --------------------------- launcher --------------------------------------

extern "C" void kernel_launch(void* const* d_in, const int* in_sizes, int n_in,
                              void* d_out, int out_size, void* d_ws, size_t ws_size,
                              hipStream_t stream) {
    const float* x   = (const float*)d_in[0];
    const int*   did = (const int*)  d_in[1];
    const float* rA  = (const float*)d_in[2];
    const float* rS  = (const float*)d_in[3];
    const float* rb  = (const float*)d_in[4];
    const float* eA  = (const float*)d_in[5];
    const float* eS  = (const float*)d_in[6];
    const float* eb  = (const float*)d_in[7];
    const float* ndW = (const float*)d_in[8];
    const float* ndb = (const float*)d_in[9];
    const float* dr  = (const float*)d_in[10];
    float* out = (float*)d_out;

    // workspace layout (bytes)
    char* ws = (char*)d_ws;
    unsigned short* xb   = (unsigned short*)(ws + 0);                 // 12,582,912
    unsigned short* We   = (unsigned short*)(ws + 12582912);          // 37,748,736
    unsigned short* ndWT = (unsigned short*)(ws + 50331648);          // 37,748,736
    float*          Wr   = (float*)        (ws + 88080384);           //     24,576
    float*          gate = (float*)        (ws + 88104960);           //    262,144
    unsigned short* H    = (unsigned short*)(ws + 88367104);          // 50,331,648
    // total ~138.7 MB

    const int threads = 256;
    k_build_wr        <<<(kE * kC + threads - 1) / threads, threads, 0, stream>>>(rA, rS, Wr);
    k_convert_f32_bf16<<<(kTok * kC + threads - 1) / threads, threads, 0, stream>>>(x, xb, kTok * kC);
    k_build_we        <<<(kE * kDFF * kC + threads - 1) / threads, threads, 0, stream>>>(eA, eS, We);
    k_transpose_ndw   <<<(kE * kC * kDFF + threads - 1) / threads, threads, 0, stream>>>(ndW, ndWT);
    k_router_gate     <<<(kTok + threads - 1) / threads, threads, 0, stream>>>(x, Wr, rb, dr, did, gate);
    k_zero_f32        <<<(kTok * kC + threads - 1) / threads, threads, 0, stream>>>(out, kTok * kC);

    // GEMM grids: block tile 64(M) x 256(N)
    dim3 g1(kDFF / 256, kTok / 64);   // (12, 128)
    dim3 g2(kC   / 256, kTok / 64);   // (3, 128)
    for (int e = 0; e < kE; ++e) {
        k_ffn1<<<g1, threads, 0, stream>>>(xb, We + (size_t)e * kDFF * kC, eb + (size_t)e * kDFF, H);
        k_ffn2<<<g2, threads, 0, stream>>>(H, ndWT + (size_t)e * kC * kDFF, ndb + (size_t)e * kC,
                                           gate, e, out);
    }
}
